// Experts_86543591014908
// MI455X (gfx1250) — matrix operand order repaired and checked by reading, not compiled, not run on hardware
//
#include <hip/hip_runtime.h>

// CDNA5 / gfx1250: wave32, WMMA bf16 16x16x32, f32 accumulate.
// Per-expert SwiGLU MLP. Roofline: 206 GFLOP vs 3.22 GB f32 weights ->
// HBM-bound. Stream weights once (non-temporal), convert with
// v_cvt_pk_bf16_f32, double-buffer LDS staging so weight-load latency hides
// behind the WMMA block, pin B-fragments so each sub-step is one ds clause +
// one wait + 8 back-to-back v_wmma, and stage 4x4 blocks so the LDS drain is
// packed ds_store_b64 with zero half-register shuffling.

typedef __attribute__((ext_vector_type(16))) __bf16 v16bf;
typedef __attribute__((ext_vector_type(8)))  __bf16 v8bf;
typedef __attribute__((ext_vector_type(4)))  __bf16 v4bf;
typedef __attribute__((ext_vector_type(8)))  float  v8f;
typedef __attribute__((ext_vector_type(4)))  float  v4f;
typedef __attribute__((ext_vector_type(8)))  unsigned v8u;

#define E_ 16
#define T_ 128
#define D_ 2048
#define H_ 8192
#define KSTEP 64
#define LDSTR 72   // LDS k-stride per column (elements); 144 B rows, 8/16B aligned

__device__ __forceinline__ v4bf cvt4(v4f v) {           // -> v_cvt_pk_bf16_f32
  return __builtin_convertvector(v, v4bf);
}
__device__ __forceinline__ unsigned short f2bf_bits(float f) {
  return __builtin_bit_cast(unsigned short, (__bf16)f);
}
__device__ __forceinline__ v4f ntload4(const float* p) {
  return __builtin_nontemporal_load((const v4f*)p);
}
__device__ __forceinline__ v16bf lds_frag(const __bf16* p) {
  v8bf lo = *(const v8bf*)(p);
  v8bf hi = *(const v8bf*)(p + 16);
  return __builtin_shufflevector(lo, hi, 0, 1, 2, 3, 4, 5, 6, 7,
                                 8, 9, 10, 11, 12, 13, 14, 15);
}
__device__ __forceinline__ v16bf make_afrag(v4f a0, v4f a1, v4f a2, v4f a3) {
  v4bf b0 = cvt4(a0), b1 = cvt4(a1), b2 = cvt4(a2), b3 = cvt4(a3);
  v8bf lo = __builtin_shufflevector(b0, b1, 0, 1, 2, 3, 4, 5, 6, 7);
  v8bf hi = __builtin_shufflevector(b2, b3, 0, 1, 2, 3, 4, 5, 6, 7);
  return __builtin_shufflevector(lo, hi, 0, 1, 2, 3, 4, 5, 6, 7,
                                 8, 9, 10, 11, 12, 13, 14, 15);
}
// Force all fragments live before the WMMA block: one scheduling wall.
__device__ __forceinline__ void pin8(v16bf* f) {
  v8u t0 = __builtin_bit_cast(v8u, f[0]), t1 = __builtin_bit_cast(v8u, f[1]);
  v8u t2 = __builtin_bit_cast(v8u, f[2]), t3 = __builtin_bit_cast(v8u, f[3]);
  v8u t4 = __builtin_bit_cast(v8u, f[4]), t5 = __builtin_bit_cast(v8u, f[5]);
  v8u t6 = __builtin_bit_cast(v8u, f[6]), t7 = __builtin_bit_cast(v8u, f[7]);
  asm volatile("" : "+v"(t0), "+v"(t1), "+v"(t2), "+v"(t3),
                    "+v"(t4), "+v"(t5), "+v"(t6), "+v"(t7));
  f[0] = __builtin_bit_cast(v16bf, t0); f[1] = __builtin_bit_cast(v16bf, t1);
  f[2] = __builtin_bit_cast(v16bf, t2); f[3] = __builtin_bit_cast(v16bf, t3);
  f[4] = __builtin_bit_cast(v16bf, t4); f[5] = __builtin_bit_cast(v16bf, t5);
  f[6] = __builtin_bit_cast(v16bf, t6); f[7] = __builtin_bit_cast(v16bf, t7);
}
__device__ __forceinline__ void pin4(v16bf* f) {
  v8u t0 = __builtin_bit_cast(v8u, f[0]), t1 = __builtin_bit_cast(v8u, f[1]);
  v8u t2 = __builtin_bit_cast(v8u, f[2]), t3 = __builtin_bit_cast(v8u, f[3]);
  asm volatile("" : "+v"(t0), "+v"(t1), "+v"(t2), "+v"(t3));
  f[0] = __builtin_bit_cast(v16bf, t0); f[1] = __builtin_bit_cast(v16bf, t1);
  f[2] = __builtin_bit_cast(v16bf, t2); f[3] = __builtin_bit_cast(v16bf, t3);
}
// Drain a thread's 4x4 staged block (4 k-rows x 4 cols) into [col][k] LDS:
// per column one packed v4bf (ds_store_b64); transpose is absorbed by cvt_pk.
__device__ __forceinline__ void drain4x4(__bf16* lds, int cg, int kq,
                                         v4f s0, v4f s1, v4f s2, v4f s3) {
  #pragma unroll
  for (int j = 0; j < 4; ++j) {
    v4f tk = {s0[j], s1[j], s2[j], s3[j]};
    *(v4bf*)&lds[(cg + j) * LDSTR + 4 * kq] = cvt4(tk);
  }
}

// ---------------------------------------------------------------------------
// Kernel 1: og[e,t,h] = (x@w_c_fc + b_fc) * silu(x@w_gate + b_gate), bf16 out.
// Grid: (H/64, E). Block: 256 (8 waves). Tile: 128 rows x 64 cols, K = D.
// ---------------------------------------------------------------------------
__global__ __launch_bounds__(256)
void swiglu_up_kernel(const float* __restrict__ x,
                      const float* __restrict__ w_c_fc,
                      const float* __restrict__ b_c_fc,
                      const float* __restrict__ w_gate,
                      const float* __restrict__ b_gate,
                      unsigned short* __restrict__ og) {
  __shared__ __align__(16) __bf16 sFc[2][64 * LDSTR];
  __shared__ __align__(16) __bf16 sGt[2][64 * LDSTR];

  const int e     = blockIdx.y;
  const int hbase = blockIdx.x * 64;
  const float* xe  = x      + (size_t)e * T_ * D_;
  const float* wf  = w_c_fc + (size_t)e * D_ * H_;
  const float* wg  = w_gate + (size_t)e * D_ * H_;
  const float* bfc = b_c_fc + (size_t)e * H_;
  const float* bgt = b_gate + (size_t)e * H_;
  unsigned short* oge = og + (size_t)e * T_ * H_;

  const int tid  = threadIdx.x;
  const int wave = tid >> 5;
  const int lane = tid & 31;
  const int nl   = lane & 15;
  const int half = lane >> 4;
  const int hk   = half * 8;    // A-frag K sub-offset (lanes 16-31 -> +8)
  const int hk2  = half * 8;    // B-frag K sub-offset (lanes 16-31 -> +8)
  const int row  = wave * 16 + nl;

  // staging map: thread owns a 4x4 block: cols cg..cg+3, k-rows 4kq..4kq+3
  const int cg = (tid & 15) * 4;
  const int kq = tid >> 4;          // 0..15

  v8f accH[4] = {};
  v8f accG[4] = {};

  // ---- prologue: stage tile 0 into buffer 0 ----
  {
    const size_t g0 = (size_t)(4 * kq) * H_ + hbase + cg;
    v4f f0 = ntload4(wf + g0), f1 = ntload4(wf + g0 + H_);
    v4f f2 = ntload4(wf + g0 + 2 * H_), f3 = ntload4(wf + g0 + 3 * H_);
    v4f g0v = ntload4(wg + g0), g1v = ntload4(wg + g0 + H_);
    v4f g2v = ntload4(wg + g0 + 2 * H_), g3v = ntload4(wg + g0 + 3 * H_);
    drain4x4(sFc[0], cg, kq, f0, f1, f2, f3);
    drain4x4(sGt[0], cg, kq, g0v, g1v, g2v, g3v);
  }

  int p = 0;
  for (int k0 = 0; k0 < D_; k0 += KSTEP) {
    __syncthreads();
    const bool has_next = (k0 + KSTEP < D_);
    const __bf16* curF = sFc[p];
    const __bf16* curG = sGt[p];
    __bf16* nxtF = sFc[p ^ 1];
    __bf16* nxtG = sGt[p ^ 1];
    const size_t gnx = (size_t)(k0 + KSTEP + 4 * kq) * H_ + hbase + cg;

    // issue next-tile w_c_fc loads now; data lands during sub-step 0 WMMAs
    v4f sF0, sF1, sF2, sF3;
    if (has_next) {
      sF0 = ntload4(wf + gnx);          sF1 = ntload4(wf + gnx + H_);
      sF2 = ntload4(wf + gnx + 2 * H_); sF3 = ntload4(wf + gnx + 3 * H_);
    }

    // ---- sub-step kk = 0 ----
    {
      const float* ap = xe + (size_t)row * D_ + k0 + hk;
      v16bf afrag = make_afrag(*(const v4f*)(ap + 0),  *(const v4f*)(ap + 4),
                               *(const v4f*)(ap + 16), *(const v4f*)(ap + 20));
      v16bf bfr[8];
      #pragma unroll
      for (int c = 0; c < 4; ++c) {
        bfr[c]     = lds_frag(&curF[(c * 16 + nl) * LDSTR + hk2]);
        bfr[c + 4] = lds_frag(&curG[(c * 16 + nl) * LDSTR + hk2]);
      }
      pin8(bfr);
      #pragma unroll
      for (int c = 0; c < 4; ++c) {
        accH[c] = __builtin_amdgcn_wmma_f32_16x16x32_bf16(
            false, afrag, false, bfr[c], (short)0, accH[c], false, false);
        accG[c] = __builtin_amdgcn_wmma_f32_16x16x32_bf16(
            false, afrag, false, bfr[c + 4], (short)0, accG[c], false, false);
      }
    }

    // drain w_c_fc staging into inactive buffer, then issue w_gate loads
    v4f sG0, sG1, sG2, sG3;
    if (has_next) {
      drain4x4(nxtF, cg, kq, sF0, sF1, sF2, sF3);
      sG0 = ntload4(wg + gnx);          sG1 = ntload4(wg + gnx + H_);
      sG2 = ntload4(wg + gnx + 2 * H_); sG3 = ntload4(wg + gnx + 3 * H_);
    }

    // ---- sub-step kk = 32 ----
    {
      const float* ap = xe + (size_t)row * D_ + k0 + 32 + hk;
      v16bf afrag = make_afrag(*(const v4f*)(ap + 0),  *(const v4f*)(ap + 4),
                               *(const v4f*)(ap + 16), *(const v4f*)(ap + 20));
      v16bf bfr[8];
      #pragma unroll
      for (int c = 0; c < 4; ++c) {
        bfr[c]     = lds_frag(&curF[(c * 16 + nl) * LDSTR + 32 + hk2]);
        bfr[c + 4] = lds_frag(&curG[(c * 16 + nl) * LDSTR + 32 + hk2]);
      }
      pin8(bfr);
      #pragma unroll
      for (int c = 0; c < 4; ++c) {
        accH[c] = __builtin_amdgcn_wmma_f32_16x16x32_bf16(
            false, afrag, false, bfr[c], (short)0, accH[c], false, false);
        accG[c] = __builtin_amdgcn_wmma_f32_16x16x32_bf16(
            false, afrag, false, bfr[c + 4], (short)0, accG[c], false, false);
      }
    }

    // drain w_gate staging
    if (has_next) {
      drain4x4(nxtG, cg, kq, sG0, sG1, sG2, sG3);
    }
    p ^= 1;
  }

  // ---- epilogue: bias + SwiGLU, store bf16 (C layout: r -> M=r / r+8) ----
  #pragma unroll
  for (int c = 0; c < 4; ++c) {
    const int col = hbase + c * 16 + nl;
    const float bh = bfc[col];
    const float bg = bgt[col];
    #pragma unroll
    for (int r = 0; r < 8; ++r) {
      float h = accH[c][r] + bh;
      float g = accG[c][r] + bg;
      float sig = 1.0f / (1.0f + __expf(-g));
      float o = h * g * sig;
      const int rr = wave * 16 + r + half * 8;
      oge[(size_t)rr * H_ + col] = f2bf_bits(o);
    }
  }
}

// ---------------------------------------------------------------------------
// Kernel 2: out[e,t,d] = og @ w_c_proj + b_c_proj (f32 out).
// Grid: (D/64, E). Block: 256. Tile: 128 rows x 64 cols, K = H.
// ---------------------------------------------------------------------------
__global__ __launch_bounds__(256)
void down_proj_kernel(const unsigned short* __restrict__ og,
                      const float* __restrict__ w_c_proj,
                      const float* __restrict__ b_c_proj,
                      float* __restrict__ out) {
  __shared__ __align__(16) __bf16 sW[2][64 * LDSTR];

  const int e     = blockIdx.y;
  const int dbase = blockIdx.x * 64;
  const unsigned short* oge = og + (size_t)e * T_ * H_;
  const float* wp = w_c_proj + (size_t)e * H_ * D_;
  const float* bp = b_c_proj + (size_t)e * D_;
  float* oute = out + (size_t)e * T_ * D_;

  const int tid  = threadIdx.x;
  const int wave = tid >> 5;
  const int lane = tid & 31;
  const int nl   = lane & 15;
  const int half = lane >> 4;
  const int hk   = half * 8;
  const int hk2  = half * 8;
  const int row  = wave * 16 + nl;

  const int cg = (tid & 15) * 4;
  const int kq = tid >> 4;

  v8f acc[4] = {};

  // ---- prologue: stage tile 0 into buffer 0 ----
  {
    const size_t g0 = (size_t)(4 * kq) * D_ + dbase + cg;
    v4f s0 = ntload4(wp + g0), s1 = ntload4(wp + g0 + D_);
    v4f s2 = ntload4(wp + g0 + 2 * D_), s3 = ntload4(wp + g0 + 3 * D_);
    drain4x4(sW[0], cg, kq, s0, s1, s2, s3);
  }

  int p = 0;
  for (int k0 = 0; k0 < H_; k0 += KSTEP) {
    __syncthreads();
    const bool has_next = (k0 + KSTEP < H_);
    const __bf16* cur = sW[p];
    __bf16* nxt = sW[p ^ 1];
    const size_t gnx = (size_t)(k0 + KSTEP + 4 * kq) * D_ + dbase + cg;

    v4f s0, s1, s2, s3;
    if (has_next) {
      s0 = ntload4(wp + gnx);          s1 = ntload4(wp + gnx + D_);
      s2 = ntload4(wp + gnx + 2 * D_); s3 = ntload4(wp + gnx + 3 * D_);
    }

    // ---- sub-step kk = 0 ----
    {
      const unsigned short* ap = oge + (size_t)row * H_ + k0 + hk;
      v8bf alo = *(const v8bf*)(ap);
      v8bf ahi = *(const v8bf*)(ap + 16);
      v16bf afrag = __builtin_shufflevector(alo, ahi, 0, 1, 2, 3, 4, 5, 6, 7,
                                            8, 9, 10, 11, 12, 13, 14, 15);
      v16bf bfr[4];
      #pragma unroll
      for (int c = 0; c < 4; ++c)
        bfr[c] = lds_frag(&cur[(c * 16 + nl) * LDSTR + hk2]);
      pin4(bfr);
      #pragma unroll
      for (int c = 0; c < 4; ++c)
        acc[c] = __builtin_amdgcn_wmma_f32_16x16x32_bf16(
            false, afrag, false, bfr[c], (short)0, acc[c], false, false);
    }

    // drain staging into inactive buffer
    if (has_next) {
      drain4x4(nxt, cg, kq, s0, s1, s2, s3);
    }

    // ---- sub-step kk = 32 ----
    {
      const unsigned short* ap = oge + (size_t)row * H_ + k0 + 32 + hk;
      v8bf alo = *(const v8bf*)(ap);
      v8bf ahi = *(const v8bf*)(ap + 16);
      v16bf afrag = __builtin_shufflevector(alo, ahi, 0, 1, 2, 3, 4, 5, 6, 7,
                                            8, 9, 10, 11, 12, 13, 14, 15);
      v16bf bfr[4];
      #pragma unroll
      for (int c = 0; c < 4; ++c)
        bfr[c] = lds_frag(&cur[(c * 16 + nl) * LDSTR + 32 + hk2]);
      pin4(bfr);
      #pragma unroll
      for (int c = 0; c < 4; ++c)
        acc[c] = __builtin_amdgcn_wmma_f32_16x16x32_bf16(
            false, afrag, false, bfr[c], (short)0, acc[c], false, false);
    }
    p ^= 1;
  }

  #pragma unroll
  for (int c = 0; c < 4; ++c) {
    const int col = dbase + c * 16 + nl;
    const float bias = bp[col];
    #pragma unroll
    for (int r = 0; r < 8; ++r) {
      const int rr = wave * 16 + r + half * 8;
      oute[(size_t)rr * D_ + col] = acc[c][r] + bias;
    }
  }
}

extern "C" void kernel_launch(void* const* d_in, const int* in_sizes, int n_in,
                              void* d_out, int out_size, void* d_ws, size_t ws_size,
                              hipStream_t stream) {
  const float* x        = (const float*)d_in[0];
  const float* w_c_fc   = (const float*)d_in[1];
  const float* b_c_fc   = (const float*)d_in[2];
  const float* w_gate   = (const float*)d_in[3];
  const float* b_gate   = (const float*)d_in[4];
  const float* w_c_proj = (const float*)d_in[5];
  const float* b_c_proj = (const float*)d_in[6];
  float* out = (float*)d_out;

  // workspace: og[E,T,H] in bf16 = 32 MB
  unsigned short* og = (unsigned short*)d_ws;

  dim3 gridA(H_ / 64, E_);
  swiglu_up_kernel<<<gridA, 256, 0, stream>>>(x, w_c_fc, b_c_fc, w_gate, b_gate, og);

  dim3 gridB(D_ / 64, E_);
  down_proj_kernel<<<gridB, 256, 0, stream>>>(og, w_c_proj, b_c_proj, out);
}
